// LSTMDecoder_18571438587964
// MI455X (gfx1250) — matrix operand order repaired and checked by reading, not compiled, not run on hardware
//
#include <hip/hip_runtime.h>

typedef __attribute__((ext_vector_type(16))) __bf16 v16bf;
typedef __attribute__((ext_vector_type(8)))  __bf16 v8bf;
typedef __attribute__((ext_vector_type(8)))  float  v8f;

#define B_   1024
#define L_   128
#define H_   256
#define OUT_ 64
#define T_   256
#define G4H  1024   // 4*H
#define BT   16     // batch rows per workgroup
#define NTHR 512    // 16 waves

// ---------------- fast activation helpers (1 exp2 + 1 rcp each) ----------------
__device__ __forceinline__ float fexp2f(float x) { return __builtin_amdgcn_exp2f(x); }
__device__ __forceinline__ float frcpf(float x)  { return __builtin_amdgcn_rcpf(x); }
__device__ __forceinline__ float sigmf(float x) {
  return frcpf(1.0f + fexp2f(-1.44269504088896f * x));
}
__device__ __forceinline__ float tanhf_(float x) {
  // tanh(x) = 1 - 2/(exp(2x)+1)
  return 1.0f - 2.0f * frcpf(1.0f + fexp2f(2.88539008177793f * x));
}

// ---------------- prep kernels ----------------
__global__ void cvt_bf16_kernel(const float* __restrict__ src, __bf16* __restrict__ dst, int n) {
  int i = blockIdx.x * blockDim.x + threadIdx.x;
  if (i < n) dst[i] = (__bf16)src[i];
}

__global__ void add_bias_kernel(const float* __restrict__ a, const float* __restrict__ b,
                                float* __restrict__ o, int n) {
  int i = blockIdx.x * blockDim.x + threadIdx.x;
  if (i < n) o[i] = a[i] + b[i];
}

// One block per batch row: h0/c0 for both layers + time-constant layer-0 input
// projection gx0 = latent @ w_ih0^T + b_ih0 + b_hh0.
__global__ __launch_bounds__(256) void init_states_kernel(
    const float* __restrict__ latent,
    const float* __restrict__ w_lh, const float* __restrict__ b_lh,
    const float* __restrict__ w_lc, const float* __restrict__ b_lc,
    const float* __restrict__ w_ih0, const float* __restrict__ b_ih0, const float* __restrict__ b_hh0,
    __bf16* __restrict__ hA, __bf16* __restrict__ hB,
    float* __restrict__ cA, float* __restrict__ cB,
    float* __restrict__ gx0) {
  __shared__ float lat[L_];
  const int b = blockIdx.x, tid = threadIdx.x;
  if (tid < L_) lat[tid] = latent[b * L_ + tid];
  __syncthreads();
  for (int col = tid; col < 2048; col += 256) {
    const float* wrow; float bias;
    if (col < 512)        { wrow = w_lh + col * L_;                 bias = b_lh[col]; }
    else if (col < 1024)  { int j = col - 512;  wrow = w_lc + j * L_;  bias = b_lc[j]; }
    else                  { int j = col - 1024; wrow = w_ih0 + j * L_; bias = b_ih0[j] + b_hh0[j]; }
    float s = bias;
    #pragma unroll 8
    for (int k = 0; k < L_; ++k) s += lat[k] * wrow[k];
    if (col < 512) {
      if (col < 256) hA[b * H_ + col]         = (__bf16)s;
      else           hB[b * H_ + (col - 256)] = (__bf16)s;
    } else if (col < 1024) {
      int j = col - 512;
      if (j < 256) cA[b * H_ + j]         = s;
      else         cB[b * H_ + (j - 256)] = s;
    } else {
      gx0[b * G4H + (col - 1024)] = s;
    }
  }
}

// ---------------- WMMA fragment loaders ----------------
// A (16x32 bf16), M = batch rows 0..15 of the LDS tile:
// lane L<16:  M=L,    K = {k0..k0+7,  k0+16..k0+23}
// lane L>=16: M=L-16, K = {k0+8..k0+15, k0+24..k0+31}
__device__ __forceinline__ v16bf load_fragA(const __bf16* hlds, int k0, int lane) {
  const int row = lane & 15;
  const int kb  = k0 + (lane >> 4) * 8;
  union { v16bf v; v8bf h[2]; } u;
  u.h[0] = *(const v8bf*)(hlds + row * H_ + kb);
  u.h[1] = *(const v8bf*)(hlds + row * H_ + kb + 16);
  return u.v;
}

// B (32x16 bf16), B[k][n] = w[nbase+n][k] (w row-major [rows][H_]):
// lane L: column n = L%16, K = {k0+(L/16)*8 .. +7, k0+16+(L/16)*8 .. +7} of one w row
__device__ __forceinline__ v16bf load_fragB(const __bf16* __restrict__ wrowbase, int k0, int lane) {
  const int n  = lane & 15;
  const int kk = k0 + (lane >> 4) * 8;
  union { v16bf v; v8bf h[2]; } u;
  u.h[0] = *(const v8bf*)(wrowbase + n * H_ + kk);
  u.h[1] = *(const v8bf*)(wrowbase + n * H_ + kk + 16);
  return u.v;
}

// ---------------- fused decoder ----------------
// Grid: 64 WGs x 512 threads. Each WG owns 16 batch rows for all T steps.
// Wave w (0..15) owns hidden-column tile w (cols w*16..w*16+15) of both layers.
__global__ __launch_bounds__(NTHR, 1) void decoder_kernel(
    const __bf16* __restrict__ whh0, const __bf16* __restrict__ wih1,
    const __bf16* __restrict__ whh1, const __bf16* __restrict__ wout,
    const float*  __restrict__ b1,   const float*  __restrict__ b_out,
    const __bf16* __restrict__ hA0,  const __bf16* __restrict__ hB0,
    const float*  __restrict__ cA0,  const float*  __restrict__ cB0,
    const float*  __restrict__ gx0,  float* __restrict__ out) {
  __shared__ __bf16 hA[2][BT * H_];   // layer-0 hidden state, double buffered
  __shared__ __bf16 hB[2][BT * H_];   // layer-1 hidden state, double buffered

  const int tid   = threadIdx.x;
  const int wave  = tid >> 5;
  const int lane  = tid & 31;
  const int lh    = lane & 15;
  const int lq    = lane >> 4;        // C/D layout: elem e -> (M = e + 8*lq, N = lh)
  const int bbase = blockIdx.x * BT;
  const int nh    = wave * 16;        // hidden column base owned by this wave

  // fill LDS buffer 0 with initial hidden states
  for (int i = tid; i < BT * H_; i += NTHR) {
    hA[0][i] = hA0[bbase * H_ + i];
    hB[0][i] = hB0[bbase * H_ + i];
  }

  // persistent per-wave registers
  v8f cAr, cBr;          // cell states (C/D layout)
  v8f gx0r[4];           // constant layer-0 accumulator init (i,f,g,o)
  float b1r[4];          // layer-1 gate biases
  #pragma unroll
  for (int e = 0; e < 8; ++e) {
    const int row = bbase + e + lq * 8;
    cAr[e] = cA0[row * H_ + nh + lh];
    cBr[e] = cB0[row * H_ + nh + lh];
    #pragma unroll
    for (int g = 0; g < 4; ++g)
      gx0r[g][e] = gx0[row * G4H + g * H_ + nh + lh];
  }
  #pragma unroll
  for (int g = 0; g < 4; ++g) b1r[g] = b1[g * H_ + nh + lh];
  const float boutr = b_out[(wave & 3) * 16 + lh];   // used by waves 0..3

  __syncthreads();

  int p = 0;
  for (int t = 0; t < T_; ++t) {
    // ================= layer 0: gates = gx0 + h0 @ w_hh0^T =================
    {
      v8f acc0 = gx0r[0], acc1 = gx0r[1], acc2 = gx0r[2], acc3 = gx0r[3];
      #pragma unroll 2
      for (int k0 = 0; k0 < H_; k0 += 32) {
        const v16bf a = load_fragA(hA[p], k0, lane);
        acc0 = __builtin_amdgcn_wmma_f32_16x16x32_bf16(false, a, false,
                 load_fragB(whh0 + (0 * H_ + nh) * H_, k0, lane), (short)0, acc0, false, false);
        acc1 = __builtin_amdgcn_wmma_f32_16x16x32_bf16(false, a, false,
                 load_fragB(whh0 + (1 * H_ + nh) * H_, k0, lane), (short)0, acc1, false, false);
        acc2 = __builtin_amdgcn_wmma_f32_16x16x32_bf16(false, a, false,
                 load_fragB(whh0 + (2 * H_ + nh) * H_, k0, lane), (short)0, acc2, false, false);
        acc3 = __builtin_amdgcn_wmma_f32_16x16x32_bf16(false, a, false,
                 load_fragB(whh0 + (3 * H_ + nh) * H_, k0, lane), (short)0, acc3, false, false);
      }
      #pragma unroll
      for (int e = 0; e < 8; ++e) {
        const float iv = sigmf(acc0[e]);
        const float fv = sigmf(acc1[e]);
        const float gv = tanhf_(acc2[e]);
        const float ov = sigmf(acc3[e]);
        const float c  = fv * cAr[e] + iv * gv;
        cAr[e] = c;
        hA[p ^ 1][(e + lq * 8) * H_ + nh + lh] = (__bf16)(ov * tanhf_(c));
      }
    }
    __syncthreads();                       // hA[p^1] = hs0_t complete

    // ====== layer 1: gates = hs0_t @ w_ih1^T + h1 @ w_hh1^T + b1 ======
    {
      v8f acc0, acc1, acc2, acc3;
      #pragma unroll
      for (int e = 0; e < 8; ++e) { acc0[e] = b1r[0]; acc1[e] = b1r[1]; acc2[e] = b1r[2]; acc3[e] = b1r[3]; }
      #pragma unroll 2
      for (int k0 = 0; k0 < H_; k0 += 32) {
        const v16bf a = load_fragA(hA[p ^ 1], k0, lane);
        acc0 = __builtin_amdgcn_wmma_f32_16x16x32_bf16(false, a, false,
                 load_fragB(wih1 + (0 * H_ + nh) * H_, k0, lane), (short)0, acc0, false, false);
        acc1 = __builtin_amdgcn_wmma_f32_16x16x32_bf16(false, a, false,
                 load_fragB(wih1 + (1 * H_ + nh) * H_, k0, lane), (short)0, acc1, false, false);
        acc2 = __builtin_amdgcn_wmma_f32_16x16x32_bf16(false, a, false,
                 load_fragB(wih1 + (2 * H_ + nh) * H_, k0, lane), (short)0, acc2, false, false);
        acc3 = __builtin_amdgcn_wmma_f32_16x16x32_bf16(false, a, false,
                 load_fragB(wih1 + (3 * H_ + nh) * H_, k0, lane), (short)0, acc3, false, false);
      }
      #pragma unroll 2
      for (int k0 = 0; k0 < H_; k0 += 32) {
        const v16bf a = load_fragA(hB[p], k0, lane);
        acc0 = __builtin_amdgcn_wmma_f32_16x16x32_bf16(false, a, false,
                 load_fragB(whh1 + (0 * H_ + nh) * H_, k0, lane), (short)0, acc0, false, false);
        acc1 = __builtin_amdgcn_wmma_f32_16x16x32_bf16(false, a, false,
                 load_fragB(whh1 + (1 * H_ + nh) * H_, k0, lane), (short)0, acc1, false, false);
        acc2 = __builtin_amdgcn_wmma_f32_16x16x32_bf16(false, a, false,
                 load_fragB(whh1 + (2 * H_ + nh) * H_, k0, lane), (short)0, acc2, false, false);
        acc3 = __builtin_amdgcn_wmma_f32_16x16x32_bf16(false, a, false,
                 load_fragB(whh1 + (3 * H_ + nh) * H_, k0, lane), (short)0, acc3, false, false);
      }
      #pragma unroll
      for (int e = 0; e < 8; ++e) {
        const float iv = sigmf(acc0[e]);
        const float fv = sigmf(acc1[e]);
        const float gv = tanhf_(acc2[e]);
        const float ov = sigmf(acc3[e]);
        const float c  = fv * cBr[e] + iv * gv;
        cBr[e] = c;
        hB[p ^ 1][(e + lq * 8) * H_ + nh + lh] = (__bf16)(ov * tanhf_(c));
      }
    }
    __syncthreads();                       // hB[p^1] = hs1_t complete

    // ================= output projection (waves 0..3) =================
    if (wave < 4) {
      v8f acc;
      #pragma unroll
      for (int e = 0; e < 8; ++e) acc[e] = boutr;
      #pragma unroll 2
      for (int k0 = 0; k0 < H_; k0 += 32) {
        const v16bf a = load_fragA(hB[p ^ 1], k0, lane);
        acc = __builtin_amdgcn_wmma_f32_16x16x32_bf16(false, a, false,
                load_fragB(wout + (wave * 16) * H_, k0, lane), (short)0, acc, false, false);
      }
      #pragma unroll
      for (int e = 0; e < 8; ++e) {
        const int brow = bbase + e + lq * 8;
        out[(brow * T_ + t) * OUT_ + wave * 16 + lh] = acc[e];
      }
    }
    p ^= 1;
  }
}

// ---------------- host launcher ----------------
extern "C" void kernel_launch(void* const* d_in, const int* in_sizes, int n_in,
                              void* d_out, int out_size, void* d_ws, size_t ws_size,
                              hipStream_t stream) {
  const float* latent = (const float*)d_in[0];
  const float* w_lh   = (const float*)d_in[1];
  const float* b_lh   = (const float*)d_in[2];
  const float* w_lc   = (const float*)d_in[3];
  const float* b_lc   = (const float*)d_in[4];
  const float* w_ih0  = (const float*)d_in[5];
  const float* w_hh0  = (const float*)d_in[6];
  const float* b_ih0  = (const float*)d_in[7];
  const float* b_hh0  = (const float*)d_in[8];
  const float* w_ih1  = (const float*)d_in[9];
  const float* w_hh1  = (const float*)d_in[10];
  const float* b_ih1  = (const float*)d_in[11];
  const float* b_hh1  = (const float*)d_in[12];
  const float* w_out  = (const float*)d_in[13];
  const float* b_out  = (const float*)d_in[14];

  char* ws = (char*)d_ws;
  size_t off = 0;
  auto alloc = [&](size_t bytes) {
    char* p = ws + off;
    off += (bytes + 255) & ~(size_t)255;
    return p;
  };
  __bf16* whh0_b = (__bf16*)alloc((size_t)G4H * H_ * 2);
  __bf16* wih1_b = (__bf16*)alloc((size_t)G4H * H_ * 2);
  __bf16* whh1_b = (__bf16*)alloc((size_t)G4H * H_ * 2);
  __bf16* wout_b = (__bf16*)alloc((size_t)OUT_ * H_ * 2);
  float*  b1     = (float*) alloc((size_t)G4H * 4);
  __bf16* hA     = (__bf16*)alloc((size_t)B_ * H_ * 2);
  __bf16* hB     = (__bf16*)alloc((size_t)B_ * H_ * 2);
  float*  cA     = (float*) alloc((size_t)B_ * H_ * 4);
  float*  cB     = (float*) alloc((size_t)B_ * H_ * 4);
  float*  gx0    = (float*) alloc((size_t)B_ * G4H * 4);

  const int nw = G4H * H_;
  cvt_bf16_kernel<<<(nw + 255) / 256, 256, 0, stream>>>(w_hh0, whh0_b, nw);
  cvt_bf16_kernel<<<(nw + 255) / 256, 256, 0, stream>>>(w_ih1, wih1_b, nw);
  cvt_bf16_kernel<<<(nw + 255) / 256, 256, 0, stream>>>(w_hh1, whh1_b, nw);
  cvt_bf16_kernel<<<(OUT_ * H_ + 255) / 256, 256, 0, stream>>>(w_out, wout_b, OUT_ * H_);
  add_bias_kernel<<<(G4H + 255) / 256, 256, 0, stream>>>(b_ih1, b_hh1, b1, G4H);

  init_states_kernel<<<B_, 256, 0, stream>>>(latent, w_lh, b_lh, w_lc, b_lc,
                                             w_ih0, b_ih0, b_hh0,
                                             hA, hB, cA, cB, gx0);

  decoder_kernel<<<B_ / BT, NTHR, 0, stream>>>(whh0_b, wih1_b, whh1_b, wout_b,
                                               b1, b_out, hA, hB, cA, cB, gx0,
                                               (float*)d_out);
}
